// AsymSearch_72567767433918
// MI455X (gfx1250) — matrix operand order repaired and checked by reading, not compiled, not run on hardware
//
#include <hip/hip_runtime.h>
#include <hip/hip_bf16.h>

typedef __attribute__((ext_vector_type(16))) _Float16 v16h;
typedef __attribute__((ext_vector_type(8)))  _Float16 v8h;
typedef __attribute__((ext_vector_type(8)))  float    v8f;

#define T_   3
#define L_   4
#define M_   3
#define B_   8
#define C_   64
#define H_   128
#define W_   128
#define OUTC 3
#define NCHUNK 18              // K = 576 = 18 chunks of 32 (9 taps x 2 cin halves)
#define NPIX (B_ * H_ * W_)    // 131072

// ---------------------------------------------------------------------------
// 1) Routing: sequential gumbel top-1 per task. Tiny scalar work, 1 wave.
// ---------------------------------------------------------------------------
__global__ void route_kernel(const float* __restrict__ alpha0,
                             const float* __restrict__ alphas,
                             const float* __restrict__ g0,
                             const float* __restrict__ gs,
                             int* __restrict__ sel, float* __restrict__ stsc) {
    int t = threadIdx.x;
    if (t >= T_) return;
    int idx = 0;
    for (int layer = 0; layer < L_; ++layer) {
        float a[M_], g[M_];
        if (layer == 0) {
            for (int m = 0; m < M_; ++m) { a[m] = alpha0[t * M_ + m]; g[m] = g0[t * M_ + m]; }
        } else {
            const float* ap = alphas + (((layer - 1) * T_ + t) * M_ + idx) * M_;
            const float* gp = gs     + (((layer - 1) * T_ + t) * M_ + idx) * M_;
            for (int m = 0; m < M_; ++m) { a[m] = ap[m]; g[m] = gp[m]; }
        }
        // log_softmax(a) + g   (TAU == 1)
        float amax = a[0];
        for (int m = 1; m < M_; ++m) amax = fmaxf(amax, a[m]);
        float se = 0.f;
        for (int m = 0; m < M_; ++m) se += expf(a[m] - amax);
        float lse = logf(se) + amax;
        float logits[M_];
        for (int m = 0; m < M_; ++m) logits[m] = (a[m] - lse) + g[m];
        // argmax (first max, like jnp.argmax); softmax is monotone so same argmax
        int ni = 0; float lmax = logits[0];
        for (int m = 1; m < M_; ++m) if (logits[m] > lmax) { lmax = logits[m]; ni = m; }
        float psum = 0.f;
        for (int m = 0; m < M_; ++m) psum += expf(logits[m] - lmax);
        float p_sel = 1.0f / psum;                 // probs[ni]
        float st = 1.0f - p_sel + p_sel;           // straight-through scale (fwd)
        sel[t * L_ + layer]  = ni;
        stsc[t * L_ + layer] = st;
        idx = ni;
    }
}

// ---------------------------------------------------------------------------
// 2) x (fp32 NCHW) -> f16 NHWC staging buffer (shared h0 for all tasks)
// ---------------------------------------------------------------------------
__global__ void nchw_to_nhwc_f16(const float* __restrict__ x, _Float16* __restrict__ out) {
    int id = blockIdx.x * blockDim.x + threadIdx.x;   // over B*H*W
    if (id >= NPIX) return;
    int xc = id % W_; int rest = id / W_;
    int y = rest % H_; int b = rest / H_;
    for (int c = 0; c < C_; ++c) {
        float v = x[(((size_t)(b * C_ + c) * H_ + y) * W_) + xc];
        out[(size_t)id * C_ + c] = (_Float16)v;
    }
}

// ---------------------------------------------------------------------------
// 3) Pack selected 3x3 weights (scaled by st) into WMMA A-fragment order.
//    Layout per (t,layer): [mtile(4)][chunk(18)][lane(32)*16 + j] halves.
//    A 16x32 f16 fragment mapping used here (contiguous k, same order as B):
//      lane = hk*16 + m ; j -> k = hk*16 + j
// ---------------------------------------------------------------------------
__global__ void pack_weights(const float* __restrict__ enc_w,
                             const int* __restrict__ sel, const float* __restrict__ stsc,
                             _Float16* __restrict__ wpack) {
    const int TOT = T_ * L_ * 4 * NCHUNK * 512;
    int id = blockIdx.x * blockDim.x + threadIdx.x;
    if (id >= TOT) return;
    int f = id;
    int j     = f & 15;  f >>= 4;
    int lane  = f & 31;  f >>= 5;
    int chunk = f % NCHUNK; f /= NCHUNK;
    int mtile = f & 3;   f >>= 2;
    int layer = f % L_;  int t = f / L_;
    int m  = lane & 15, hk = lane >> 4;
    int klocal  = (hk << 4) + j;
    int kglobal = chunk * 32 + klocal;       // k = tap*64 + cin
    int tap = kglobal >> 6, cin = kglobal & 63;
    int ky = tap / 3, kx = tap % 3;
    int cout = mtile * 16 + m;
    int s = sel[t * L_ + layer];
    float w = enc_w[(((((size_t)layer * M_ + s) * C_ + cout) * C_ + cin) * 3 + ky) * 3 + kx];
    wpack[id] = (_Float16)(w * stsc[t * L_ + layer]);
}

// ---------------------------------------------------------------------------
// 4) Implicit-GEMM 3x3 conv via WMMA. One wave: 64 couts x 16 pixels.
//    Block = 128 threads (4 waves) -> 64 consecutive x per block.
//    B 32x16 f16 fragment: lane = hk*16 + n ; j -> k = hk*16 + j
//    -> per lane: 16 consecutive channels of one (padded) neighbor pixel.
// ---------------------------------------------------------------------------
__global__ void __launch_bounds__(128)
conv_wmma(const _Float16* __restrict__ actIn, _Float16* __restrict__ actOut,
          const _Float16* __restrict__ wp, const float* __restrict__ enc_b,
          const int* __restrict__ sel, int t, int layer) {
    int lane = threadIdx.x & 31;
    int wave = threadIdx.x >> 5;
    int xt   = blockIdx.x & 1;            // W/64 == 2
    int rest = blockIdx.x >> 1;
    int y = rest % H_;
    int b = rest / H_;
    int n  = lane & 15;
    int hs = lane >> 4;
    int xpix = xt * 64 + wave * 16 + n;

    v8f acc0 = {}, acc1 = {}, acc2 = {}, acc3 = {};

    for (int c = 0; c < NCHUNK; ++c) {
        int tap = c >> 1;
        int dy = tap / 3 - 1, dx = tap % 3 - 1;
        int cin0 = ((c & 1) << 5) + (hs << 4);
        int yy = y + dy, xx = xpix + dx;
        v16h bfrag = {};
        if ((unsigned)yy < (unsigned)H_ && (unsigned)xx < (unsigned)W_) {
            bfrag = *(const v16h*)(actIn + ((((size_t)(b * H_ + yy) * W_ + xx) << 6) + cin0));
        }
        const _Float16* wb = wp + (size_t)c * 512 + lane * 16;
        v16h a0 = *(const v16h*)(wb + 0 * (size_t)NCHUNK * 512);
        v16h a1 = *(const v16h*)(wb + 1 * (size_t)NCHUNK * 512);
        v16h a2 = *(const v16h*)(wb + 2 * (size_t)NCHUNK * 512);
        v16h a3 = *(const v16h*)(wb + 3 * (size_t)NCHUNK * 512);
        acc0 = __builtin_amdgcn_wmma_f32_16x16x32_f16(false, a0, false, bfrag, (short)0, acc0, false, false);
        acc1 = __builtin_amdgcn_wmma_f32_16x16x32_f16(false, a1, false, bfrag, (short)0, acc1, false, false);
        acc2 = __builtin_amdgcn_wmma_f32_16x16x32_f16(false, a2, false, bfrag, (short)0, acc2, false, false);
        acc3 = __builtin_amdgcn_wmma_f32_16x16x32_f16(false, a3, false, bfrag, (short)0, acc3, false, false);
    }

    int s = sel[t * L_ + layer];
    const float* bb = enc_b + (size_t)(layer * M_ + s) * C_;
    size_t obase = (((size_t)(b * H_ + y) * W_ + xpix) << 6) + (hs << 3);

    // D fragment: lane -> (m = hs*8 + r, n); 8 consecutive couts per lane.
#define STORE_MT(MT, ACC)                                               \
    {                                                                   \
        int cout0 = (MT) * 16 + (hs << 3);                              \
        v8h ov;                                                         \
        for (int r = 0; r < 8; ++r) {                                   \
            float v = ACC[r] + bb[cout0 + r];                           \
            v = v > 0.f ? v : 0.f;                                      \
            ov[r] = (_Float16)v;                                        \
        }                                                               \
        *(v8h*)(actOut + obase + (MT) * 16) = ov;                       \
    }
    STORE_MT(0, acc0) STORE_MT(1, acc1) STORE_MT(2, acc2) STORE_MT(3, acc3)
#undef STORE_MT
}

// ---------------------------------------------------------------------------
// 5) Per-task 1x1 decoder (+ channel-norm for task 2), fp32 NCHW output.
// ---------------------------------------------------------------------------
__global__ void decode_kernel(const _Float16* __restrict__ act,
                              const float* __restrict__ dec_w, const float* __restrict__ dec_b,
                              float* __restrict__ out, int t) {
    int id = blockIdx.x * blockDim.x + threadIdx.x;   // over B*H*W
    if (id >= NPIX) return;
    const _Float16* ap = act + (size_t)id * C_;
    float y0 = dec_b[t * OUTC + 0];
    float y1 = dec_b[t * OUTC + 1];
    float y2 = dec_b[t * OUTC + 2];
    const float* w0 = dec_w + (size_t)(t * OUTC + 0) * C_;
    const float* w1 = dec_w + (size_t)(t * OUTC + 1) * C_;
    const float* w2 = dec_w + (size_t)(t * OUTC + 2) * C_;
    for (int c = 0; c < C_; ++c) {
        float h = (float)ap[c];
        y0 += w0[c] * h;
        y1 += w1[c] * h;
        y2 += w2[c] * h;
    }
    if (t == 2) {  // "normal" task: L2-normalize over channels (no epsilon, as reference)
        float nrm = sqrtf(y0 * y0 + y1 * y1 + y2 * y2);
        y0 /= nrm; y1 /= nrm; y2 /= nrm;
    }
    int b = id / (H_ * W_);
    int p = id % (H_ * W_);
    size_t ob = ((size_t)(t * B_ + b) * OUTC) * (size_t)(H_ * W_) + p;
    out[ob]                        = y0;
    out[ob + (size_t)(H_ * W_)]    = y1;
    out[ob + (size_t)(2 * H_ * W_)]= y2;
}

// ---------------------------------------------------------------------------
// Host launcher
// ---------------------------------------------------------------------------
extern "C" void kernel_launch(void* const* d_in, const int* in_sizes, int n_in,
                              void* d_out, int out_size, void* d_ws, size_t ws_size,
                              hipStream_t stream) {
    const float* x      = (const float*)d_in[0];
    const float* alpha0 = (const float*)d_in[1];
    const float* alphas = (const float*)d_in[2];
    const float* g0     = (const float*)d_in[3];
    const float* gs     = (const float*)d_in[4];
    const float* enc_w  = (const float*)d_in[5];
    const float* enc_b  = (const float*)d_in[6];
    const float* dec_w  = (const float*)d_in[7];
    const float* dec_b  = (const float*)d_in[8];
    float* out = (float*)d_out;

    char* ws = (char*)d_ws;
    const size_t ACT_BYTES = (size_t)NPIX * C_ * sizeof(_Float16);   // 16 MB
    int*      sel   = (int*)ws;
    float*    stsc  = (float*)(ws + 256);
    _Float16* wpack = (_Float16*)(ws + 512);                          // 884736 B
    _Float16* actX  = (_Float16*)(ws + (1u << 20));
    _Float16* actA  = (_Float16*)(ws + (1u << 20) + ACT_BYTES);
    _Float16* actB  = (_Float16*)(ws + (1u << 20) + 2 * ACT_BYTES);

    route_kernel<<<1, 32, 0, stream>>>(alpha0, alphas, g0, gs, sel, stsc);
    nchw_to_nhwc_f16<<<(NPIX + 255) / 256, 256, 0, stream>>>(x, actX);
    {
        const int TOTW = T_ * L_ * 4 * NCHUNK * 512;                  // 442368
        pack_weights<<<(TOTW + 255) / 256, 256, 0, stream>>>(enc_w, sel, stsc, wpack);
    }

    const int conv_grid = B_ * H_ * (W_ / 64);                        // 2048
    for (int t = 0; t < T_; ++t) {
        const _Float16* in = actX;
        _Float16* pp[2] = { actA, actB };
        for (int layer = 0; layer < L_; ++layer) {
            _Float16* ob = pp[layer & 1];
            const _Float16* wp = wpack + (size_t)(t * L_ + layer) * 4 * NCHUNK * 512;
            conv_wmma<<<conv_grid, 128, 0, stream>>>(in, ob, wp, enc_b, sel, t, layer);
            in = ob;
        }
        decode_kernel<<<(NPIX + 255) / 256, 256, 0, stream>>>(in, dec_w, dec_b, out, t);
    }
}
